// EncoderRNNFeed_30107720745675
// MI455X (gfx1250) — matrix operand order repaired and checked by reading, not compiled, not run on hardware
//
#include <hip/hip_runtime.h>
#include <hip/hip_bf16.h>
#include <math.h>

// Problem sizes
#define BB   32
#define TT   512
#define EE   256
#define HH   512
#define G3   1536   // 3*H

typedef __attribute__((ext_vector_type(16))) __bf16 v16bf;
typedef __attribute__((ext_vector_type(8)))  __bf16 v8bf;
typedef __attribute__((ext_vector_type(8)))  float  v8f;

// ---------------------------------------------------------------------------
// Fragment loaders following CDNA5 WMMA VGPR layouts (CDNA5 ISA):
//  A (16x32 bf16, MxK): lane l -> m = l&15 ; hi = l>>4
//    elems e in [0,8):  K = k0 + 8*hi + e            (contiguous 16B)
//    elems e in [8,16): K = k0 + 16 + 8*hi + (e-8)   (contiguous 16B)
//  B (32x16 bf16, KxN): lane l -> n = l&15 ; hi = l>>4
//    elems e in [0,8):  K = k0 + 8*hi + e ; elems e in [8,16): K = k0 + 16 + 8*hi + (e-8)
//  C/D (16x16 f32): lane l -> n = l&15 ; vgpr r -> m = r + 8*(l>>4)
// ---------------------------------------------------------------------------

// A from global bf16, row-major [M, ld]
__device__ inline v16bf load_a_frag_g(const __bf16* base, int ld, int m0, int k0, int lane) {
    int m  = m0 + (lane & 15);
    int hi = lane >> 4;
    const __bf16* p0 = base + (size_t)m * ld + k0 + 8 * hi;
    v16bf a;
#pragma unroll
    for (int e = 0; e < 8; ++e) a[e] = p0[e];
#pragma unroll
    for (int e = 0; e < 8; ++e) a[8 + e] = p0[16 + e];
    return a;
}

// A from LDS bf16 shadow of h, row-major [M, ld]; two ds_load_b128 per lane
__device__ inline v16bf load_a_frag_ldsb(const __bf16* sh, int ld, int m0, int k0, int lane) {
    int m  = m0 + (lane & 15);
    int hi = lane >> 4;
    const __bf16* p = sh + m * ld + k0 + 8 * hi;
    v8bf lo = *(const v8bf*)p;
    v8bf hb = *(const v8bf*)(p + 16);
    v16bf a;
#pragma unroll
    for (int e = 0; e < 8; ++e) { a[e] = lo[e]; a[8 + e] = hb[e]; }
    return a;
}

// B(k,n) = W[n0+n][k0+k] with W row-major [N, ld]  (GEMM computes X @ W^T)
__device__ inline v16bf load_b_frag_g(const __bf16* base, int ld, int n0, int k0, int lane) {
    int n  = n0 + (lane & 15);
    int hi = lane >> 4;
    const __bf16* p = base + (size_t)n * ld + k0 + 8 * hi;
    v16bf b;
#pragma unroll
    for (int e = 0; e < 8; ++e) { b[e] = p[e]; b[8 + e] = p[16 + e]; }
    return b;
}

#define WMMA_BF16(a, b, c) \
    __builtin_amdgcn_wmma_f32_16x16x32_bf16(false, (a), false, (b), (short)0, (c), false, false)

// ---------------------------------------------------------------------------
// K1: convert w_ih [1536,256] and w_hh [1536,512] f32 -> bf16
// ---------------------------------------------------------------------------
__global__ void cvt_weights_kernel(const float* __restrict__ wih,
                                   const float* __restrict__ whh,
                                   __bf16* __restrict__ wih_o,
                                   __bf16* __restrict__ whh_o) {
    int idx = blockIdx.x * blockDim.x + threadIdx.x;
    const int n1 = G3 * EE;          // 393216
    const int n2 = G3 * HH;          // 786432
    if (idx < n1) {
        wih_o[idx] = (__bf16)wih[idx];
    } else if (idx < n1 + n2) {
        int j = idx - n1;
        whh_o[j] = (__bf16)whh[j];
    }
}

// ---------------------------------------------------------------------------
// K2: embedding gather + cast (float4 loads, 4 bf16 per thread)
// ---------------------------------------------------------------------------
__global__ void embed_gather_kernel(const int* __restrict__ tok,
                                    const float* __restrict__ table,
                                    __bf16* __restrict__ out) {
    int idx = blockIdx.x * blockDim.x + threadIdx.x;   // over 16384*64
    int row = idx >> 6;
    int c4  = (idx & 63) * 4;
    int t   = tok[row];
    const float4 v = *(const float4*)(table + (size_t)t * EE + c4);
    __bf16* o = out + (size_t)row * EE + c4;
    o[0] = (__bf16)v.x; o[1] = (__bf16)v.y; o[2] = (__bf16)v.z; o[3] = (__bf16)v.w;
}

// ---------------------------------------------------------------------------
// K3: x_proj[m, g] = sum_e emb[m,e] * w_ih[g,e] + b_ih[g]
//     M=16384, N=1536, K=256. Each wave -> 32x32 output tile (2x2 WMMA frags).
// ---------------------------------------------------------------------------
__global__ __launch_bounds__(256) void gemm_xproj_kernel(const __bf16* __restrict__ A,
                                                         const __bf16* __restrict__ W,
                                                         const float*  __restrict__ bias,
                                                         float* __restrict__ C) {
    int gtid = blockIdx.x * blockDim.x + threadIdx.x;
    int wid  = gtid >> 5;
    int lane = gtid & 31;
    const int ntiles_n = G3 / 32;                 // 48
    int tm0 = (wid / ntiles_n) * 32;
    int tn0 = (wid % ntiles_n) * 32;
    if (tm0 >= BB * TT) return;

    v8f acc00 = {}, acc01 = {}, acc10 = {}, acc11 = {};
#pragma unroll
    for (int kk = 0; kk < EE; kk += 32) {
        v16bf a0 = load_a_frag_g(A, EE, tm0,      kk, lane);
        v16bf a1 = load_a_frag_g(A, EE, tm0 + 16, kk, lane);
        v16bf b0 = load_b_frag_g(W, EE, tn0,      kk, lane);
        v16bf b1 = load_b_frag_g(W, EE, tn0 + 16, kk, lane);
        acc00 = WMMA_BF16(a0, b0, acc00);
        acc01 = WMMA_BF16(a0, b1, acc01);
        acc10 = WMMA_BF16(a1, b0, acc10);
        acc11 = WMMA_BF16(a1, b1, acc11);
    }

    int nlo   = lane & 15;
    int mhalf = (lane >> 4) * 8;
    v8f* accs[2][2] = {{&acc00, &acc01}, {&acc10, &acc11}};
#pragma unroll
    for (int mi = 0; mi < 2; ++mi) {
#pragma unroll
        for (int ni = 0; ni < 2; ++ni) {
            int n    = tn0 + ni * 16 + nlo;
            float bv = bias[n];
            v8f& ac  = *accs[mi][ni];
#pragma unroll
            for (int r = 0; r < 8; ++r) {
                int m = tm0 + mi * 16 + mhalf + r;
                C[(size_t)m * G3 + n] = ac[r] + bv;
            }
        }
    }
}

// ---------------------------------------------------------------------------
// K4: persistent single-workgroup GRU recurrence (32 wave32 waves, one WGP).
//     LDS: h f32 (66KB) + h bf16 shadow (34KB) + hp f32 (198KB)  ~= 298KB.
//     Per step:
//       - prefetch this step's strided x_proj block (1536 cachelines) so it
//         lands in WGP$/L2 while the GEMM runs (global_prefetch_b8)
//       - hp = h @ w_hh^T via WMMA; each wave owns 3 N-tiles x 2 M-tiles so
//         every w_hh byte is loaded exactly once per step (1.5MB/step)
//       - fused gate math in f32, update h (f32 + bf16 shadow)
// ---------------------------------------------------------------------------
__global__ __launch_bounds__(1024) void gru_recurrent_kernel(
        const float*  __restrict__ xproj,   // [B*T, 3H]
        const float*  __restrict__ h0,      // [B, H]
        const __bf16* __restrict__ whh,     // [3H, H] bf16
        const float*  __restrict__ b_hh,    // [3H]
        float* __restrict__ out,            // [B, T, H]
        float* __restrict__ hlast)          // [B, H]
{
    __shared__ float sh_h [BB][HH + 8];
    __shared__ __align__(16) __bf16 sh_hb[BB][HH + 16];
    __shared__ float sh_hp[BB][G3 + 8];

    const int tid  = threadIdx.x;
    const int lane = tid & 31;
    const int wave = tid >> 5;

    // init h from h0 (f32 state + bf16 shadow)
    for (int i = tid; i < BB * HH; i += 1024) {
        float v = h0[i];
        sh_h [i >> 9][i & 511] = v;
        sh_hb[i >> 9][i & 511] = (__bf16)v;
    }
    __syncthreads();

    // each wave: 3 N-tiles (of 16 cols) x both M-tiles; 32 waves cover 96 N-tiles
    const int tn_base = wave * 3;

    for (int t = 0; t < TT; ++t) {
        // ---- prefetch x_proj block for this step (read after the barrier) ----
        {
            // wave index doubles as batch row b = 0..31; 48 lines of 128B per row
            const char* xrow = (const char*)(xproj + ((size_t)(wave * TT + t)) * G3);
            __builtin_prefetch(xrow + lane * 128, 0, 3);
            if (lane < 16)
                __builtin_prefetch(xrow + (32 + lane) * 128, 0, 3);
        }

        // ---- hp = h @ whh^T : M=32, N=1536, K=512 ----
        v8f acc[2][3] = {};
#pragma unroll 4
        for (int kk = 0; kk < HH; kk += 32) {
            v16bf a0 = load_a_frag_ldsb(&sh_hb[0][0], HH + 16, 0,  kk, lane);
            v16bf a1 = load_a_frag_ldsb(&sh_hb[0][0], HH + 16, 16, kk, lane);
#pragma unroll
            for (int tt = 0; tt < 3; ++tt) {
                v16bf b = load_b_frag_g(whh, HH, (tn_base + tt) * 16, kk, lane);
                acc[0][tt] = WMMA_BF16(a0, b, acc[0][tt]);
                acc[1][tt] = WMMA_BF16(a1, b, acc[1][tt]);
            }
        }
        // write hp tiles to LDS
        {
            int nlo = lane & 15;
            int mh  = (lane >> 4) * 8;
#pragma unroll
            for (int mi = 0; mi < 2; ++mi) {
#pragma unroll
                for (int tt = 0; tt < 3; ++tt) {
                    int n = (tn_base + tt) * 16 + nlo;
#pragma unroll
                    for (int r = 0; r < 8; ++r)
                        sh_hp[mi * 16 + mh + r][n] = acc[mi][tt][r];
                }
            }
        }
        __syncthreads();

        // ---- fused gates: r|z|n ordering, h update ----
        for (int i = tid; i < BB * HH; i += 1024) {
            int b = i >> 9;
            int j = i & 511;
            const float* xp = xproj + ((size_t)(b * TT + t)) * G3;
            float xr = xp[j];
            float xz = xp[HH + j];
            float xn = xp[2 * HH + j];
            float hr = sh_hp[b][j]          + b_hh[j];
            float hz = sh_hp[b][HH + j]     + b_hh[HH + j];
            float hn = sh_hp[b][2 * HH + j] + b_hh[2 * HH + j];
            float r = 1.0f / (1.0f + expf(-(xr + hr)));
            float z = 1.0f / (1.0f + expf(-(xz + hz)));
            float n = tanhf(xn + r * hn);
            float hold = sh_h[b][j];
            float hnew = (1.0f - z) * n + z * hold;
            out[((size_t)b * TT + t) * HH + j] = hnew;
            sh_h [b][j] = hnew;
            sh_hb[b][j] = (__bf16)hnew;
        }
        __syncthreads();
    }

    for (int i = tid; i < BB * HH; i += 1024)
        hlast[i] = sh_h[i >> 9][i & 511];
}

// ---------------------------------------------------------------------------
// Launch
// ---------------------------------------------------------------------------
extern "C" void kernel_launch(void* const* d_in, const int* in_sizes, int n_in,
                              void* d_out, int out_size, void* d_ws, size_t ws_size,
                              hipStream_t stream) {
    const int*   input     = (const int*)  d_in[0];   // [B,T]
    const float* hidden    = (const float*)d_in[1];   // [1,B,H]
    const float* emb_table = (const float*)d_in[3];   // [V,E]
    const float* w_ih      = (const float*)d_in[4];   // [3H,E]
    const float* w_hh      = (const float*)d_in[5];   // [3H,H]
    const float* b_ih      = (const float*)d_in[6];   // [3H]
    const float* b_hh      = (const float*)d_in[7];   // [3H]

    float* out   = (float*)d_out;                     // [B,T,H]
    float* hlast = out + (size_t)BB * TT * HH;        // [1,B,H]

    // workspace layout (all 256B aligned)
    char* ws = (char*)d_ws;
    __bf16* emb_bf  = (__bf16*)(ws);                              //  8,388,608 B
    __bf16* wih_bf  = (__bf16*)(ws + 8388608);                    //    786,432 B
    __bf16* whh_bf  = (__bf16*)(ws + 8388608 + 786432);           //  1,572,864 B
    float*  xproj   = (float*) (ws + 8388608 + 786432 + 1572864); // 100,663,296 B

    // K1: weight conversion
    {
        int total  = G3 * EE + G3 * HH;                // 1,179,648
        int blocks = (total + 255) / 256;
        cvt_weights_kernel<<<blocks, 256, 0, stream>>>(w_ih, w_hh, wih_bf, whh_bf);
    }
    // K2: embedding gather (4 elems per thread)
    {
        int total = BB * TT * (EE / 4);                // 1,048,576
        embed_gather_kernel<<<total / 256, 256, 0, stream>>>(input, emb_table, emb_bf);
    }
    // K3: x_proj GEMM  (24576 waves = 3072 blocks of 8 waves)
    {
        int nwaves = (BB * TT / 32) * (G3 / 32);       // 512*48 = 24576
        gemm_xproj_kernel<<<nwaves / 8, 256, 0, stream>>>(emb_bf, wih_bf, b_ih, xproj);
    }
    // K4: persistent recurrence (single workgroup, 32 waves)
    {
        gru_recurrent_kernel<<<1, 1024, 0, stream>>>(xproj, hidden, whh_bf, b_hh, out, hlast);
    }
}
